// MegaTransformerSelfAttention_89928025244603
// MI455X (gfx1250) — matrix operand order repaired and checked by reading, not compiled, not run on hardware
//
#include <hip/hip_runtime.h>
#include <hip/hip_bf16.h>

// MegaTransformerSelfAttention for MI455X (gfx1250), bf16 WMMA + fp32 accum.
// B=2, S=2048, D=1024, H=16, DH=64; ALiBi before 1/sqrt(d), tanh softcap 30,
// causal, flash-style streaming softmax (never materializes SxS scores).
// V is stored transposed per head ([B][H][DH][S]) so every WMMA operand load
// in the attention inner loop is a contiguous 128-bit vector load.

#define B_  2
#define S_  2048
#define D_  1024
#define H_  16
#define DH_ 64

typedef __bf16 bf16;
typedef __bf16 v16bf __attribute__((ext_vector_type(16)));
typedef __bf16 v8bf  __attribute__((ext_vector_type(8)));
typedef float  v8f   __attribute__((ext_vector_type(8)));

#define WMMA_BF16(A, Bm, C) \
  __builtin_amdgcn_wmma_f32_16x16x32_bf16(false, (A), false, (Bm), (short)0, (C), false, false)

__device__ __forceinline__ int lane_id() { return (int)(threadIdx.x & 31u); }

// A-fragment (16x32, bf16, row-major source with leading dim ld):
// lanes 0-15: row M=lane, elems 0..15 -> K=kk..kk+15
// lanes 16-31: row M=lane-16, elems 0..15 -> K=kk+16..kk+31
__device__ __forceinline__ v16bf load_a_frag(const bf16* base, int ld, int m0, int kk) {
  const int lane = lane_id();
  const int m = m0 + (lane & 15);
  const int k = kk + ((lane >> 4) << 4);
  const bf16* p = base + (size_t)m * ld + k;
  union { v16bf v; v8bf h[2]; } u;
  u.h[0] = *(const v8bf*)(p);
  u.h[1] = *(const v8bf*)(p + 8);
  return u.v;
}

// B-fragment (32x16) where B[k][n] = W[n][k] (W row-major, ld): contiguous run.
// lanes 0-15: N=lane, K=kk+e; lanes 16-31: N=lane-16, K=kk+16+e
__device__ __forceinline__ v16bf load_bT_frag(const bf16* W, int ld, int n0, int kk) {
  const int lane = lane_id();
  const int n = n0 + (lane & 15);
  const int k = kk + ((lane >> 4) << 4);
  return *(const v16bf*)(W + (size_t)n * ld + k);
}

// ---------------- elementwise f32 -> bf16 ----------------
__global__ void k_f32_to_bf16(const float* __restrict__ src, bf16* __restrict__ dst, int n) {
  int i = blockIdx.x * blockDim.x + threadIdx.x;
  const int stride = gridDim.x * blockDim.x;
  for (; i < n; i += stride) dst[i] = (bf16)src[i];
}

// ---------------- QKV projection: [4096,1024] @ W^T, head-split output ----------------
// One wave per 16x64 output tile. mat 0/1 = Q/K -> [B][H][S][DH]; mat 2 = V -> [B][H][DH][S].
__global__ void k_qkv_gemm(const bf16* __restrict__ X,
                           const bf16* __restrict__ Wq, const bf16* __restrict__ Wk,
                           const bf16* __restrict__ Wv,
                           bf16* __restrict__ Q, bf16* __restrict__ K, bf16* __restrict__ Vt) {
  const int gw = blockIdx.x * (blockDim.x >> 5) + (threadIdx.x >> 5);
  const int per_mat = (B_ * S_ / 16) * (D_ / 64);   // 4096
  const int mat = gw / per_mat;
  const int rem = gw % per_mat;
  const int m0 = (rem / (D_ / 64)) * 16;
  const int n0 = (rem % (D_ / 64)) * 64;
  const bf16* W = (mat == 0) ? Wq : (mat == 1) ? Wk : Wv;

  v8f acc[4] = {};
  for (int kk = 0; kk < D_; kk += 32) {
    v16bf a = load_a_frag(X, D_, m0, kk);
#pragma unroll
    for (int s = 0; s < 4; ++s) {
      v16bf bm = load_bT_frag(W, D_, n0 + s * 16, kk);
      acc[s] = WMMA_BF16(a, bm, acc[s]);
    }
  }
  const int lane = lane_id();
  const int mbase = (lane >> 4) << 3;   // C-layout: upper lanes hold rows 8..15
  const int ncol = lane & 15;
#pragma unroll
  for (int s = 0; s < 4; ++s) {
#pragma unroll
    for (int r = 0; r < 8; ++r) {
      const int m = m0 + mbase + r;
      const int n = n0 + s * 16 + ncol;
      const int b = m >> 11, sq = m & (S_ - 1);
      const int h = n >> 6, d = n & 63;
      const size_t bh = (size_t)b * H_ + h;
      const bf16 val = (bf16)acc[s][r];
      if (mat == 0)      Q [(bh * S_ + sq) * DH_ + d] = val;
      else if (mat == 1) K [(bh * S_ + sq) * DH_ + d] = val;
      else               Vt[(bh * DH_ + d) * S_ + sq] = val;   // transposed per head
    }
  }
}

// ---------------- flash attention: one wave per (b,h, 16-query tile) ----------------
__global__ void k_flash_attn(const bf16* __restrict__ Q, const bf16* __restrict__ K,
                             const bf16* __restrict__ Vt, bf16* __restrict__ CTX) {
  __shared__ bf16 lds_p[8][16 * 32];                 // per-wave 16x32 prob tile
  const int wave = (int)(threadIdx.x >> 5);
  const int gw = blockIdx.x * (blockDim.x >> 5) + wave;
  const int qtiles = S_ / 16;                        // 128
  const int bh = gw / qtiles;
  const int qt = gw % qtiles;
  const int b = bh / H_, h = bh % H_;
  const float slope = exp2f(-0.5f * (float)(h + 1)); // 2^(-(h+1)*8/H)

  const bf16* Qh  = Q  + (size_t)bh * S_ * DH_;
  const bf16* Kh  = K  + (size_t)bh * S_ * DH_;
  const bf16* Vth = Vt + (size_t)bh * DH_ * S_;      // [DH][S]

  const int lane = lane_id();
  const int mbase = (lane >> 4) << 3;
  const int ncol = lane & 15;
  const int q0 = qt * 16;

  const v16bf aq0 = load_a_frag(Qh, DH_, q0, 0);
  const v16bf aq1 = load_a_frag(Qh, DH_, q0, 32);

  v8f o[4] = {};
  float mrow[8], lrow[8];
#pragma unroll
  for (int r = 0; r < 8; ++r) { mrow[r] = -3.0e38f; lrow[r] = 0.0f; }

  for (int ks = 0; ks < q0 + 16; ks += 32) {         // causal: keys <= q0+15
    v8f s0 = {}, s1 = {};
    {
      v16bf b00 = load_bT_frag(Kh, DH_, ks, 0);      // B[k][n] = K[key][dh]
      v16bf b01 = load_bT_frag(Kh, DH_, ks, 32);
      s0 = WMMA_BF16(aq0, b00, s0);
      s0 = WMMA_BF16(aq1, b01, s0);
      v16bf b10 = load_bT_frag(Kh, DH_, ks + 16, 0);
      v16bf b11 = load_bT_frag(Kh, DH_, ks + 16, 32);
      s1 = WMMA_BF16(aq0, b10, s1);
      s1 = WMMA_BF16(aq1, b11, s1);
    }
    float p0[8], p1[8], mnew[8];
#pragma unroll
    for (int r = 0; r < 8; ++r) {
      const int q = q0 + mbase + r;
      const int k0i = ks + ncol;
      const int k1i = ks + 16 + ncol;
      float v0 = (s0[r] - slope * (float)(q - k0i)) * 0.125f;   // (+alibi)/sqrt(64)
      v0 = 30.0f * tanhf(v0 * (1.0f / 30.0f));                  // softcap
      if (k0i > q) v0 = -3.0e38f;                               // causal
      float v1 = (s1[r] - slope * (float)(q - k1i)) * 0.125f;
      v1 = 30.0f * tanhf(v1 * (1.0f / 30.0f));
      if (k1i > q) v1 = -3.0e38f;
      p0[r] = v0; p1[r] = v1;
    }
    float t_red[8];
#pragma unroll
    for (int r = 0; r < 8; ++r) {                    // row max over 16-lane half
      float t = fmaxf(p0[r], p1[r]);
#pragma unroll
      for (int x = 1; x < 16; x <<= 1) t = fmaxf(t, __shfl_xor(t, x, 32));
      mnew[r] = fmaxf(mrow[r], t);
    }
#pragma unroll
    for (int r = 0; r < 8; ++r) {
      const float alpha = __expf(mrow[r] - mnew[r]);
      const float e0 = __expf(p0[r] - mnew[r]);
      const float e1 = __expf(p1[r] - mnew[r]);
      float t = e0 + e1;
#pragma unroll
      for (int x = 1; x < 16; x <<= 1) t += __shfl_xor(t, x, 32);
      t_red[r] = t;
      lrow[r] = lrow[r] * alpha + t;
      mrow[r] = mnew[r];
#pragma unroll
      for (int s = 0; s < 4; ++s) o[s][r] *= alpha;
      const int m = mbase + r;                       // C-layout -> [m][k] in LDS
      lds_p[wave][m * 32 + ncol] = (bf16)e0;
      lds_p[wave][m * 32 + 16 + ncol] = (bf16)e1;
    }
    (void)t_red;
    // per-wave LDS region; same-wave DS ops are in-order (DScnt handled by compiler)
    const v16bf ap = load_a_frag(&lds_p[wave][0], 32, 0, 0);   // P as A-frag (16x32)
#pragma unroll
    for (int s = 0; s < 4; ++s) {
      // B[k][n] = V[key=k][dh=n] = Vt[n][k]  -> contiguous b128 load
      v16bf bv = load_bT_frag(Vth, S_, s * 16, ks);
      o[s] = WMMA_BF16(ap, bv, o[s]);
    }
  }
  // normalize and write ctx as [B][S][H*DH] bf16
#pragma unroll
  for (int s = 0; s < 4; ++s) {
#pragma unroll
    for (int r = 0; r < 8; ++r) {
      const int q = q0 + mbase + r;
      const float val = o[s][r] / lrow[r];
      CTX[(((size_t)b * S_) + q) * D_ + h * DH_ + s * 16 + ncol] = (bf16)val;
    }
  }
}

// ---------------- output projection: ctx @ Wo^T + bo, fp32 out ----------------
__global__ void k_out_gemm(const bf16* __restrict__ CTXb, const bf16* __restrict__ Wo,
                           const float* __restrict__ bo, float* __restrict__ OUT) {
  const int gw = blockIdx.x * (blockDim.x >> 5) + (threadIdx.x >> 5);
  const int m0 = (gw / (D_ / 64)) * 16;
  const int n0 = (gw % (D_ / 64)) * 64;
  v8f acc[4] = {};
  for (int kk = 0; kk < D_; kk += 32) {
    v16bf a = load_a_frag(CTXb, D_, m0, kk);
#pragma unroll
    for (int s = 0; s < 4; ++s) {
      v16bf bm = load_bT_frag(Wo, D_, n0 + s * 16, kk);
      acc[s] = WMMA_BF16(a, bm, acc[s]);
    }
  }
  const int lane = lane_id();
  const int mbase = (lane >> 4) << 3;
  const int ncol = lane & 15;
#pragma unroll
  for (int s = 0; s < 4; ++s) {
    const int n = n0 + s * 16 + ncol;
    const float bias = bo[n];
#pragma unroll
    for (int r = 0; r < 8; ++r) {
      OUT[(size_t)(m0 + mbase + r) * D_ + n] = acc[s][r] + bias;
    }
  }
}

extern "C" void kernel_launch(void* const* d_in, const int* in_sizes, int n_in,
                              void* d_out, int out_size, void* d_ws, size_t ws_size,
                              hipStream_t stream) {
  const float* hs = (const float*)d_in[0];
  const float* Wq = (const float*)d_in[1];
  const float* Wk = (const float*)d_in[2];
  const float* Wv = (const float*)d_in[3];
  const float* Wo = (const float*)d_in[4];
  const float* bo = (const float*)d_in[5];
  float* out = (float*)d_out;

  char* ws = (char*)d_ws;
  size_t off = 0;
  bf16* Xbf = (bf16*)(ws + off); off += (size_t)B_ * S_ * D_ * 2;
  bf16* Wqb = (bf16*)(ws + off); off += (size_t)D_ * D_ * 2;
  bf16* Wkb = (bf16*)(ws + off); off += (size_t)D_ * D_ * 2;
  bf16* Wvb = (bf16*)(ws + off); off += (size_t)D_ * D_ * 2;
  bf16* Wob = (bf16*)(ws + off); off += (size_t)D_ * D_ * 2;
  bf16* Qb  = (bf16*)(ws + off); off += (size_t)B_ * H_ * S_ * DH_ * 2;
  bf16* Kb  = (bf16*)(ws + off); off += (size_t)B_ * H_ * S_ * DH_ * 2;
  bf16* Vtb = (bf16*)(ws + off); off += (size_t)B_ * H_ * S_ * DH_ * 2;
  bf16* Cb  = (bf16*)(ws + off); off += (size_t)B_ * S_ * D_ * 2;
  (void)ws_size; (void)in_sizes; (void)n_in; (void)out_size;

  k_f32_to_bf16<<<1024, 256, 0, stream>>>(hs, Xbf, B_ * S_ * D_);
  k_f32_to_bf16<<<512, 256, 0, stream>>>(Wq, Wqb, D_ * D_);
  k_f32_to_bf16<<<512, 256, 0, stream>>>(Wk, Wkb, D_ * D_);
  k_f32_to_bf16<<<512, 256, 0, stream>>>(Wv, Wvb, D_ * D_);
  k_f32_to_bf16<<<512, 256, 0, stream>>>(Wo, Wob, D_ * D_);

  // 3 matrices * 256 m-tiles * 16 n-tiles = 12288 waves; 8 waves/block
  k_qkv_gemm<<<1536, 256, 0, stream>>>(Xbf, Wqb, Wkb, Wvb, Qb, Kb, Vtb);
  // B*H*(S/16) = 4096 waves
  k_flash_attn<<<512, 256, 0, stream>>>(Qb, Kb, Vtb, Cb);
  // 256 * 16 = 4096 waves
  k_out_gemm<<<512, 256, 0, stream>>>(Cb, Wob, bo, out);
}
